// VMoE_53480932770320
// MI455X (gfx1250) — compile-verified, matrix operand order checked by reading, not run on hardware
//
#include <hip/hip_runtime.h>
#include <hip/hip_bf16.h>
#include <math.h>

typedef __bf16 bf16;
typedef __attribute__((ext_vector_type(16))) __bf16 v16bf;
typedef __attribute__((ext_vector_type(8)))  __bf16 bf16x8;
typedef __attribute__((ext_vector_type(8)))  float   v8f;
typedef __attribute__((ext_vector_type(4)))  float   f32x4;

#define BB    8
#define DD    768
#define NSEQ  197
#define TT    1576          // BB*NSEQ
#define TP    1600          // padded token rows
#define NHH   12
#define HDD   64
#define NPD   208           // padded seq (13*16)
#define KPD   224           // padded key dim for P@V (7*32)
#define DHH   3072
#define NEXP  8
#define NPAT  196

// ---------------- WMMA helpers (layouts per CDNA5 ISA 7.12.2) ----------------
__device__ __forceinline__ v8f wmma_bf16f32(v16bf a, v16bf b, v8f c) {
  return __builtin_amdgcn_wmma_f32_16x16x32_bf16(false, a, false, b, (short)0, c,
                                                 false, false);
}

// per-lane fragment: p already includes row base + (lane>>4)*8; chunks at +0,+16
__device__ __forceinline__ v16bf frag_bf16(const bf16* p) {
  bf16x8 lo = *(const bf16x8*)(p);
  bf16x8 hi = *(const bf16x8*)(p + 16);
  v16bf v;
#pragma unroll
  for (int i = 0; i < 8; ++i) { v[i] = lo[i]; v[i + 8] = hi[i]; }
  return v;
}

// same fragment but sourced from fp32 weights (convert to bf16 in-register)
__device__ __forceinline__ v16bf frag_f32(const float* p) {
  f32x4 a0 = *(const f32x4*)(p);
  f32x4 a1 = *(const f32x4*)(p + 4);
  f32x4 a2 = *(const f32x4*)(p + 16);
  f32x4 a3 = *(const f32x4*)(p + 20);
  v16bf v;
#pragma unroll
  for (int i = 0; i < 4; ++i) {
    v[i]      = (__bf16)a0[i];
    v[i + 4]  = (__bf16)a1[i];
    v[i + 8]  = (__bf16)a2[i];
    v[i + 12] = (__bf16)a3[i];
  }
  return v;
}

// ---------------- generic linear: C = act(A[M,K]bf16 @ W[N,K]^T + bias) -----
// 32(M) x 64(N) tile per wave: each converted W fragment feeds 2 WMMAs.
// EPI: 0 store bf16, 1 store f32 (scatter rows), 2 f32 +=, 3 f32 += slotw*val
template <int EPI, bool GELU>
__global__ __launch_bounds__(32) void linear_kernel(
    const bf16* __restrict__ A, int lda,
    const float* __restrict__ W, long long wStride,
    const float* __restrict__ bias, int bStride,
    void* __restrict__ Cv, int ldc, long long cStride,
    int M, int K,
    const int* __restrict__ toks, const int* __restrict__ cnts,
    const float* __restrict__ slotw, int listBase,
    const int* __restrict__ rowCidx)
{
  const int g = blockIdx.z;
  const int listIdx = listBase + g;
  const int Meff = cnts ? cnts[listIdx] : M;
  const int mbase = blockIdx.x * 32;
  if (mbase >= Meff) return;
  const int* tl = toks ? toks + (size_t)listIdx * TP : nullptr;
  const float* Wg = W + (size_t)(g & 7) * (size_t)wStride;
  const float* bp = bias ? bias + (size_t)(g & 7) * (size_t)bStride : nullptr;
  char* Cb = (char*)Cv + (size_t)(g >> 3) * (size_t)cStride;

  const int l  = threadIdx.x;
  const int r  = l & 15;
  const int hi = l >> 4;

  int am0 = mbase + r;      if (am0 > Meff - 1) am0 = Meff - 1;
  int am1 = mbase + 16 + r; if (am1 > Meff - 1) am1 = Meff - 1;
  const int arow0 = tl ? tl[am0] : am0;
  const int arow1 = tl ? tl[am1] : am1;
  const bf16* ap0 = A + (size_t)arow0 * lda + hi * 8;
  const bf16* ap1 = A + (size_t)arow1 * lda + hi * 8;

  const int ncol = blockIdx.y * 64;
  const float* wp0 = Wg + (size_t)(ncol +  0 + r) * K + hi * 8;
  const float* wp1 = Wg + (size_t)(ncol + 16 + r) * K + hi * 8;
  const float* wp2 = Wg + (size_t)(ncol + 32 + r) * K + hi * 8;
  const float* wp3 = Wg + (size_t)(ncol + 48 + r) * K + hi * 8;

  v8f c00 = {}, c01 = {}, c02 = {}, c03 = {};
  v8f c10 = {}, c11 = {}, c12 = {}, c13 = {};
  for (int k0 = 0; k0 < K; k0 += 32) {
    v16bf a0 = frag_bf16(ap0 + k0);
    v16bf a1 = frag_bf16(ap1 + k0);
    v16bf b0 = frag_f32(wp0 + k0);
    c00 = wmma_bf16f32(a0, b0, c00);
    c10 = wmma_bf16f32(a1, b0, c10);
    v16bf b1 = frag_f32(wp1 + k0);
    c01 = wmma_bf16f32(a0, b1, c01);
    c11 = wmma_bf16f32(a1, b1, c11);
    v16bf b2 = frag_f32(wp2 + k0);
    c02 = wmma_bf16f32(a0, b2, c02);
    c12 = wmma_bf16f32(a1, b2, c12);
    v16bf b3 = frag_f32(wp3 + k0);
    c03 = wmma_bf16f32(a0, b3, c03);
    c13 = wmma_bf16f32(a1, b3, c13);
  }
  v8f accs[2][4] = {{c00, c01, c02, c03}, {c10, c11, c12, c13}};
#pragma unroll
  for (int half = 0; half < 2; ++half) {
#pragma unroll
    for (int sub = 0; sub < 4; ++sub) {
      const int n = ncol + sub * 16 + r;
      const float bv = bp ? bp[n] : 0.0f;
#pragma unroll
      for (int e = 0; e < 8; ++e) {
        const int m = mbase + half * 16 + hi * 8 + e;
        if (m >= Meff) continue;
        float v = accs[half][sub][e] + bv;
        if (GELU) v = 0.5f * v * (1.0f + erff(v * 0.70710678118654752f));
        const int orow = tl ? tl[m] : (rowCidx ? rowCidx[m] : m);
        if (EPI == 0) {
          ((bf16*)Cb)[(size_t)orow * ldc + n] = (bf16)v;
        } else if (EPI == 1) {
          ((float*)Cb)[(size_t)orow * ldc + n] = v;
        } else if (EPI == 2) {
          float* p = (float*)Cb + (size_t)orow * ldc + n;
          *p = *p + v;
        } else {
          const float sw = slotw[(size_t)listIdx * TP + m];
          float* p = (float*)Cb + (size_t)orow * ldc + n;
          *p = *p + sw * v;
        }
      }
    }
  }
}

// ---------------- LayerNorm: z(f32) -> xb(bf16), pad rows zeroed ------------
__device__ __forceinline__ float blockReduceSum(float v, float* sm) {
  __syncthreads();
#pragma unroll
  for (int o = 16; o > 0; o >>= 1) v += __shfl_down(v, o, 32);
  const int wid = threadIdx.x >> 5;
  if ((threadIdx.x & 31) == 0) sm[wid] = v;
  __syncthreads();
  if (threadIdx.x == 0) {
    float t = 0.0f;
#pragma unroll
    for (int i = 0; i < 8; ++i) t += sm[i];
    sm[0] = t;
  }
  __syncthreads();
  return sm[0];
}

__global__ __launch_bounds__(256) void ln_kernel(
    const float* __restrict__ z, const float* __restrict__ w,
    const float* __restrict__ b, bf16* __restrict__ out, int Mvalid)
{
  const int row = blockIdx.x;
  bf16* orow = out + (size_t)row * DD;
  if (row >= Mvalid) {
    for (int d = threadIdx.x; d < DD; d += 256) orow[d] = (bf16)0.0f;
    return;
  }
  const float* x = z + (size_t)row * DD;
  const int d = threadIdx.x;
  const float v0 = x[d], v1 = x[d + 256], v2 = x[d + 512];
  __shared__ float sm[8];
  const float mean = blockReduceSum(v0 + v1 + v2, sm) * (1.0f / 768.0f);
  const float d0 = v0 - mean, d1 = v1 - mean, d2 = v2 - mean;
  const float var = blockReduceSum(d0 * d0 + d1 * d1 + d2 * d2, sm) * (1.0f / 768.0f);
  const float rinv = rsqrtf(var + 1e-5f);
  orow[d]       = (bf16)(d0 * rinv * w[d]       + b[d]);
  orow[d + 256] = (bf16)(d1 * rinv * w[d + 256] + b[d + 256]);
  orow[d + 512] = (bf16)(d2 * rinv * w[d + 512] + b[d + 512]);
}

// ---------------- patchify + embedding assembly -----------------------------
__global__ void patchify_kernel(const float* __restrict__ x, bf16* __restrict__ pat) {
  const long long i = (long long)blockIdx.x * 256 + threadIdx.x;
  if (i >= (long long)BB * NPAT * DD) return;
  const int kk = (int)(i % DD);
  const int row = (int)(i / DD);
  const int b = row / NPAT, pidx = row % NPAT;
  const int gy = pidx / 14, gx = pidx % 14;
  const int c = kk / 256, rem = kk % 256, py = rem / 16, px = rem % 16;
  pat[i] = (bf16)x[((size_t)(b * 3 + c) * 224 + gy * 16 + py) * 224 + gx * 16 + px];
}

__global__ void patch_rows_kernel(int* __restrict__ rc) {
  const int i = blockIdx.x * 256 + threadIdx.x;
  if (i < BB * NPAT) rc[i] = (i / NPAT) * NSEQ + 1 + (i % NPAT);
}

__global__ void poscls_kernel(float* __restrict__ z, const float* __restrict__ cls,
                              const float* __restrict__ pos) {
  const long long i = (long long)blockIdx.x * 256 + threadIdx.x;
  if (i >= (long long)TT * DD) return;
  const int d = (int)(i % DD);
  const int t = (int)(i / DD);
  const int s = t % NSEQ;
  if (s == 0) z[i] = cls[d] + pos[d];
  else        z[i] += pos[(size_t)s * DD + d];
}

// ---------------- attention -------------------------------------------------
// S[bh][208][208] = Q tile (16x64) @ K^T, bf16 WMMA, f32 out
__global__ __launch_bounds__(32) void attn_score_kernel(
    const bf16* __restrict__ qkvb, float* __restrict__ sbuf)
{
  const int qt = blockIdx.x, kt = blockIdx.y, bh = blockIdx.z;
  const int b = bh / NHH, h = bh % NHH;
  const int l = threadIdx.x, r = l & 15, hi = l >> 4;
  int qs = qt * 16 + r; if (qs > NSEQ - 1) qs = NSEQ - 1;
  int ks = kt * 16 + r; if (ks > NSEQ - 1) ks = NSEQ - 1;
  const bf16* qp = qkvb + (size_t)(b * NSEQ + qs) * 2304 + h * HDD + hi * 8;
  const bf16* kp = qkvb + (size_t)(b * NSEQ + ks) * 2304 + DD + h * HDD + hi * 8;
  v8f acc = {};
#pragma unroll
  for (int k0 = 0; k0 < HDD; k0 += 32)
    acc = wmma_bf16f32(frag_bf16(qp + k0), frag_bf16(kp + k0), acc);
  float* srow = sbuf + (size_t)bh * NPD * NPD;
#pragma unroll
  for (int e = 0; e < 8; ++e)
    srow[(size_t)(qt * 16 + hi * 8 + e) * NPD + kt * 16 + r] = acc[e];
}

__global__ __launch_bounds__(32) void softmax_kernel(
    const float* __restrict__ sbuf, bf16* __restrict__ pbuf)
{
  const int s = blockIdx.x, bh = blockIdx.y, l = threadIdx.x;
  bf16* prow = pbuf + ((size_t)bh * NPD + s) * KPD;
  if (s >= NSEQ) {
    for (int c = l; c < KPD; c += 32) prow[c] = (bf16)0.0f;
    return;
  }
  const float* srow = sbuf + ((size_t)bh * NPD + s) * NPD;
  float v[7];
  float mx = -1e30f;
#pragma unroll
  for (int i = 0; i < 7; ++i) {
    const int c = l + 32 * i;
    const float t = (c < NSEQ) ? srow[c] * 0.125f : -1e30f;
    v[i] = t; mx = fmaxf(mx, t);
  }
#pragma unroll
  for (int o = 16; o > 0; o >>= 1) mx = fmaxf(mx, __shfl_xor(mx, o, 32));
  float sum = 0.0f;
#pragma unroll
  for (int i = 0; i < 7; ++i) {
    const float e = (v[i] > -1e29f) ? __expf(v[i] - mx) : 0.0f;
    v[i] = e; sum += e;
  }
#pragma unroll
  for (int o = 16; o > 0; o >>= 1) sum += __shfl_xor(sum, o, 32);
  const float rs = 1.0f / sum;
#pragma unroll
  for (int i = 0; i < 7; ++i) {
    const int c = l + 32 * i;
    prow[c] = (bf16)((c < NSEQ) ? v[i] * rs : 0.0f);
  }
}

// V -> vT[bh][64][224] (zero-padded keys) so P@V B-fragments are contiguous
__global__ void vtrans_kernel(const bf16* __restrict__ qkvb, bf16* __restrict__ vT) {
  const long long i = (long long)blockIdx.x * 256 + threadIdx.x;
  if (i >= (long long)BB * NHH * HDD * KPD) return;
  const int key = (int)(i % KPD);
  const int d   = (int)((i / KPD) % HDD);
  const int bh  = (int)(i / ((long long)KPD * HDD));
  const int b = bh / NHH, h = bh % NHH;
  bf16 val = (bf16)0.0f;
  if (key < NSEQ)
    val = qkvb[(size_t)(b * NSEQ + key) * 2304 + 2 * DD + h * HDD + d];
  vT[i] = val;
}

// O = P @ V  -> abuf[t][h*64+d] bf16
__global__ __launch_bounds__(32) void attn_out_kernel(
    const bf16* __restrict__ pbuf, const bf16* __restrict__ vT,
    bf16* __restrict__ abuf)
{
  const int qt = blockIdx.x, bh = blockIdx.y;
  const int b = bh / NHH, h = bh % NHH;
  const int l = threadIdx.x, r = l & 15, hi = l >> 4;
  const bf16* ap = pbuf + ((size_t)bh * NPD + qt * 16 + r) * KPD + hi * 8;
  const bf16* vp0 = vT + ((size_t)bh * HDD +  0 + r) * KPD + hi * 8;
  const bf16* vp1 = vT + ((size_t)bh * HDD + 16 + r) * KPD + hi * 8;
  const bf16* vp2 = vT + ((size_t)bh * HDD + 32 + r) * KPD + hi * 8;
  const bf16* vp3 = vT + ((size_t)bh * HDD + 48 + r) * KPD + hi * 8;
  v8f a0 = {}, a1 = {}, a2 = {}, a3 = {};
  for (int k0 = 0; k0 < KPD; k0 += 32) {
    v16bf a = frag_bf16(ap + k0);
    a0 = wmma_bf16f32(a, frag_bf16(vp0 + k0), a0);
    a1 = wmma_bf16f32(a, frag_bf16(vp1 + k0), a1);
    a2 = wmma_bf16f32(a, frag_bf16(vp2 + k0), a2);
    a3 = wmma_bf16f32(a, frag_bf16(vp3 + k0), a3);
  }
  v8f accs[4] = {a0, a1, a2, a3};
#pragma unroll
  for (int sub = 0; sub < 4; ++sub) {
#pragma unroll
    for (int e = 0; e < 8; ++e) {
      const int s = qt * 16 + hi * 8 + e;
      if (s < NSEQ)
        abuf[(size_t)(b * NSEQ + s) * DD + h * HDD + sub * 16 + r] =
            (bf16)accs[sub][e];
    }
  }
}

// ---------------- MoE gate: top-2 + softmax + per-(slot,expert) lists -------
__global__ void zero_cnt_kernel(int* __restrict__ c) {
  if (threadIdx.x < 16) c[threadIdx.x] = 0;
}

__global__ __launch_bounds__(32) void gate_kernel(
    const bf16* __restrict__ xb, const float* __restrict__ gw,
    const float* __restrict__ gb, int* __restrict__ cnt,
    int* __restrict__ toks, float* __restrict__ slotw)
{
  const int t = blockIdx.x, l = threadIdx.x;
  const bf16* x = xb + (size_t)t * DD;
  float xv[24];
#pragma unroll
  for (int i = 0; i < 24; ++i) xv[i] = (float)x[l + 32 * i];
  float lg[NEXP];
#pragma unroll
  for (int e = 0; e < NEXP; ++e) {
    const float* w = gw + (size_t)e * DD;
    float s = 0.0f;
#pragma unroll
    for (int i = 0; i < 24; ++i) s += xv[i] * w[l + 32 * i];
#pragma unroll
    for (int o = 16; o > 0; o >>= 1) s += __shfl_xor(s, o, 32);
    lg[e] = s + gb[e];
  }
  if (l == 0) {
    int i0 = 0; float v0 = lg[0];
#pragma unroll
    for (int e = 1; e < NEXP; ++e) if (lg[e] > v0) { v0 = lg[e]; i0 = e; }
    int i1 = -1; float v1 = -1e30f;
#pragma unroll
    for (int e = 0; e < NEXP; ++e)
      if (e != i0 && lg[e] > v1) { v1 = lg[e]; i1 = e; }
    const float e1 = __expf(v1 - v0);
    const float sc0 = 1.0f / (1.0f + e1);
    const float sc1 = 1.0f - sc0;
    int p0 = atomicAdd(&cnt[0 * NEXP + i0], 1);
    toks[(size_t)(0 * NEXP + i0) * TP + p0] = t;
    slotw[(size_t)(0 * NEXP + i0) * TP + p0] = sc0;
    int p1 = atomicAdd(&cnt[1 * NEXP + i1], 1);
    toks[(size_t)(1 * NEXP + i1) * TP + p1] = t;
    slotw[(size_t)(1 * NEXP + i1) * TP + p1] = sc1;
  }
}

// ---------------- classifier head ------------------------------------------
__global__ __launch_bounds__(256) void head_kernel(
    const bf16* __restrict__ xb, const float* __restrict__ hw,
    const float* __restrict__ hb, float* __restrict__ out)
{
  const int c = blockIdx.x * 256 + threadIdx.x;
  const int b = blockIdx.y;
  if (c >= 1000) return;
  const bf16* x = xb + (size_t)(b * NSEQ) * DD;
  const float* w = hw + (size_t)c * DD;
  float s = 0.0f;
  for (int d = 0; d < DD; ++d) s += (float)x[d] * w[d];
  out[b * 1000 + c] = s + hb[c];
}

// ============================ host launcher =================================
extern "C" void kernel_launch(void* const* d_in, const int* in_sizes, int n_in,
                              void* d_out, int out_size, void* d_ws, size_t ws_size,
                              hipStream_t stream) {
  (void)in_sizes; (void)n_in; (void)out_size;
  const float* x      = (const float*)d_in[0];
  const float* patchw = (const float*)d_in[1];
  const float* patchb = (const float*)d_in[2];
  const float* clstok = (const float*)d_in[3];
  const float* pose   = (const float*)d_in[4];
  const float* ln1w   = (const float*)d_in[5];
  const float* ln1b   = (const float*)d_in[6];
  const float* qkvw   = (const float*)d_in[7];
  const float* qkvbb  = (const float*)d_in[8];
  const float* projw  = (const float*)d_in[9];
  const float* projb  = (const float*)d_in[10];
  const float* ln2w   = (const float*)d_in[11];
  const float* ln2b   = (const float*)d_in[12];
  const float* fc1w   = (const float*)d_in[13];
  const float* fc1b   = (const float*)d_in[14];
  const float* fc2w   = (const float*)d_in[15];
  const float* fc2b   = (const float*)d_in[16];
  const float* gatew  = (const float*)d_in[17];
  const float* gateb  = (const float*)d_in[18];
  const float* ew1    = (const float*)d_in[19];
  const float* eb1    = (const float*)d_in[20];
  const float* ew2    = (const float*)d_in[21];
  const float* eb2    = (const float*)d_in[22];
  const float* normw  = (const float*)d_in[23];
  const float* normb  = (const float*)d_in[24];
  const float* headw  = (const float*)d_in[25];
  const float* headb  = (const float*)d_in[26];

  char* w = (char*)d_ws;
  size_t off = 0;
  auto alloc = [&](size_t bytes) -> char* {
    char* p = w + off;
    off = (off + bytes + 255) & ~(size_t)255;
    return p;
  };
  bf16*  s_pat  = (bf16*)alloc((size_t)BB * NPAT * DD * 2);
  bf16*  s_xb   = (bf16*)alloc((size_t)TP * DD * 2);
  float* s_z    = (float*)alloc((size_t)TT * DD * 4);
  bf16*  s_qkv  = (bf16*)alloc((size_t)TP * 3 * DD * 2);
  float* s_s    = (float*)alloc((size_t)BB * NHH * NPD * NPD * 4);
  bf16*  s_p    = (bf16*)alloc((size_t)BB * NHH * NPD * KPD * 2);
  bf16*  s_vt   = (bf16*)alloc((size_t)BB * NHH * HDD * KPD * 2);
  bf16*  s_a    = (bf16*)alloc((size_t)TP * DD * 2);
  bf16*  s_h    = (bf16*)alloc((size_t)2 * TP * DHH * 2);
  int*   s_toks = (int*)alloc((size_t)16 * TP * 4);
  float* s_sw   = (float*)alloc((size_t)16 * TP * 4);
  int*   s_cnt  = (int*)alloc(64);
  int*   s_rc   = (int*)alloc((size_t)BB * NPAT * 4);
  if (off > ws_size) return;  // workspace too small; bail

  const dim3 blk32(32), blk256(256);

  // ---- patch embedding ----
  patchify_kernel<<<(BB * NPAT * DD + 255) / 256, blk256, 0, stream>>>(x, s_pat);
  patch_rows_kernel<<<(BB * NPAT + 255) / 256, blk256, 0, stream>>>(s_rc);
  linear_kernel<1, false><<<dim3(49, 12, 1), blk32, 0, stream>>>(
      s_pat, DD, patchw, 0, patchb, 0, s_z, DD, 0, BB * NPAT, DD,
      nullptr, nullptr, nullptr, 0, s_rc);
  poscls_kernel<<<((long long)TT * DD + 255) / 256, blk256, 0, stream>>>(
      s_z, clstok, pose);

  int di = 0, mi = 0;
  for (int i = 0; i < 12; ++i) {
    // attention block
    ln_kernel<<<TP, blk256, 0, stream>>>(s_z, ln1w + i * DD, ln1b + i * DD, s_xb, TT);
    linear_kernel<0, false><<<dim3(50, 36, 1), blk32, 0, stream>>>(
        s_xb, DD, qkvw + (size_t)i * 3 * DD * DD, 0, qkvbb + (size_t)i * 3 * DD, 0,
        s_qkv, 3 * DD, 0, TT, DD, nullptr, nullptr, nullptr, 0, nullptr);
    attn_score_kernel<<<dim3(13, 13, BB * NHH), blk32, 0, stream>>>(s_qkv, s_s);
    softmax_kernel<<<dim3(NPD, BB * NHH), blk32, 0, stream>>>(s_s, s_p);
    vtrans_kernel<<<((long long)BB * NHH * HDD * KPD + 255) / 256, blk256, 0,
                    stream>>>(s_qkv, s_vt);
    attn_out_kernel<<<dim3(13, BB * NHH), blk32, 0, stream>>>(s_p, s_vt, s_a);
    linear_kernel<2, false><<<dim3(50, 12, 1), blk32, 0, stream>>>(
        s_a, DD, projw + (size_t)i * DD * DD, 0, projb + (size_t)i * DD, 0,
        s_z, DD, 0, TT, DD, nullptr, nullptr, nullptr, 0, nullptr);

    // MLP / MoE block
    ln_kernel<<<TP, blk256, 0, stream>>>(s_z, ln2w + i * DD, ln2b + i * DD, s_xb, TT);
    if (i & 1) {  // MoE layer (positions 1,3,5,7,9,11)
      zero_cnt_kernel<<<1, blk32, 0, stream>>>(s_cnt);
      gate_kernel<<<TT, blk32, 0, stream>>>(
          s_xb, gatew + (size_t)mi * NEXP * DD, gateb + (size_t)mi * NEXP,
          s_cnt, s_toks, s_sw);
      // fc1 for both slots (k) x 8 experts in one launch; h[k][token][3072]
      linear_kernel<0, true><<<dim3(50, 48, 16), blk32, 0, stream>>>(
          s_xb, DD, ew1 + (size_t)mi * NEXP * DHH * DD, (long long)DHH * DD,
          eb1 + (size_t)mi * NEXP * DHH, DHH,
          s_h, DHH, (long long)TP * DHH * 2, TT, DD,
          s_toks, s_cnt, nullptr, 0, nullptr);
      // fc2 per slot (serialized -> race-free residual adds)
      for (int k = 0; k < 2; ++k) {
        linear_kernel<3, false><<<dim3(50, 12, 8), blk32, 0, stream>>>(
            s_h + (size_t)k * TP * DHH, DHH,
            ew2 + (size_t)mi * NEXP * DD * DHH, (long long)DD * DHH,
            eb2 + (size_t)mi * NEXP * DD, DD,
            s_z, DD, 0, TT, DHH, s_toks, s_cnt, s_sw, k * NEXP, nullptr);
      }
      ++mi;
    } else {      // dense MLP
      linear_kernel<0, true><<<dim3(50, 48, 1), blk32, 0, stream>>>(
          s_xb, DD, fc1w + (size_t)di * DHH * DD, 0, fc1b + (size_t)di * DHH, 0,
          s_h, DHH, 0, TT, DD, nullptr, nullptr, nullptr, 0, nullptr);
      linear_kernel<2, false><<<dim3(50, 12, 1), blk32, 0, stream>>>(
          s_h, DHH, fc2w + (size_t)di * DD * DHH, 0, fc2b + (size_t)di * DD, 0,
          s_z, DD, 0, TT, DHH, nullptr, nullptr, nullptr, 0, nullptr);
      ++di;
    }
  }

  // final norm + head (cls rows)
  ln_kernel<<<TP, blk256, 0, stream>>>(s_z, normw, normb, s_xb, TT);
  head_kernel<<<dim3(4, BB), blk256, 0, stream>>>(s_xb, headw, headb, (float*)d_out);
}
